// GroupMoELayer_49314814492979
// MI455X (gfx1250) — matrix operand order repaired and checked by reading, not compiled, not run on hardware
//
#include <hip/hip_runtime.h>
#include <hip/hip_bf16.h>
#include <math.h>

typedef __attribute__((ext_vector_type(16))) __bf16 bf16x16;
typedef __attribute__((ext_vector_type(8)))  __bf16 bf16x8;
typedef __attribute__((ext_vector_type(4)))  __bf16 bf16x4;
typedef __attribute__((ext_vector_type(8)))  float  f32x8;
typedef __attribute__((ext_vector_type(4)))  float  f32x4;
typedef __attribute__((__vector_size__(16))) int    v4i;   // builtin's pointee type
typedef __attribute__((address_space(3)))    v4i    v4i_lds;

#define TILE_M 128
#define TILE_N 128
#define TILE_K 32
#define LDS_K  40   // padded LDS row stride in bf16: 80B (16B-aligned), 20 banks -> conflict-free b128 reads

// CDNA5 async global->LDS copy path (ASYNCcnt-tracked), with sync fallback.
#if defined(__has_builtin)
#  if __has_builtin(__builtin_amdgcn_global_load_async_to_lds_b128)
#    define HAVE_ASYNC_LDS 1
#  endif
#endif
#ifndef HAVE_ASYNC_LDS
#  define HAVE_ASYNC_LDS 0
#endif

static __device__ __forceinline__ void wait_async_then_barrier() {
#if HAVE_ASYNC_LDS
#  if __has_builtin(__builtin_amdgcn_s_wait_asynccnt)
  __builtin_amdgcn_s_wait_asynccnt(0);
#  else
  asm volatile("s_wait_asynccnt 0" ::: "memory");
#  endif
#endif
  __syncthreads();
}

// ---------------------------------------------------------------------------
// fp32 -> bf16 cast (vectorized)
// ---------------------------------------------------------------------------
__global__ __launch_bounds__(256) void moe_cast_bf16(const float* __restrict__ in,
                                                     __bf16* __restrict__ out, int n4) {
  int i = blockIdx.x * blockDim.x + threadIdx.x;
  if (i >= n4) return;
  f32x4 v = *(const f32x4*)(in + (size_t)i * 4);
  bf16x4 o;
  o[0] = (__bf16)v[0]; o[1] = (__bf16)v[1]; o[2] = (__bf16)v[2]; o[3] = (__bf16)v[3];
  *(bf16x4*)(out + (size_t)i * 4) = o;
}

// ---------------------------------------------------------------------------
// Router: group argmax + per-group top-2 expert softmax -> coefficients c[N][G*E]
// one wave32 per token
// ---------------------------------------------------------------------------
__device__ inline float wave_sum(float v) {
#pragma unroll
  for (int off = 16; off > 0; off >>= 1) v += __shfl_xor(v, off, 32);
  return v;
}

__global__ __launch_bounds__(256) void moe_router(
    const float* __restrict__ x, const float* __restrict__ bn,
    const float* __restrict__ Wlg, const float* __restrict__ blg,
    const float* __restrict__ Wg,          // (G*E, D)
    float* __restrict__ c,                 // (N, G*E)
    int N, int D, int G, int E) {
  const int wave = threadIdx.x >> 5;
  const int lane = threadIdx.x & 31;
  const int n = blockIdx.x * (blockDim.x >> 5) + wave;
  if (n >= N) return;

  const int GE = G * E;  // == 8 here
  const float* xr  = x  + (size_t)n * D;
  const float* bnr = bn + (size_t)n * D;

  float alg[2]; alg[0] = 0.f; alg[1] = 0.f;
  float ae[8];
#pragma unroll
  for (int k = 0; k < 8; ++k) ae[k] = 0.f;

  for (int d = lane; d < D; d += 32) {
    float xv = xr[d], bv = bnr[d];
    for (int g = 0; g < G; ++g)  alg[g] += bv * Wlg[(size_t)g * D + d];
    for (int k = 0; k < GE; ++k) ae[k]  += xv * Wg[(size_t)k * D + d];
  }
  for (int g = 0; g < G; ++g)  alg[g] = wave_sum(alg[g]);
  for (int k = 0; k < GE; ++k) ae[k]  = wave_sum(ae[k]);

  if (lane == 0) {
    // argmax over groups (softmax is monotone; first-max on ties like jnp.argmax)
    int gsel = 0; float best = alg[0] + blg[0];
    for (int g = 1; g < G; ++g) {
      float r = alg[g] + blg[g];
      if (r > best) { best = r; gsel = g; }
    }
    // top-2 experts within selected group (first index on ties, like lax.top_k)
    int i1 = 0; float l1 = ae[gsel * E];
    for (int e = 1; e < E; ++e) {
      float v = ae[gsel * E + e];
      if (v > l1) { l1 = v; i1 = e; }
    }
    int i2 = 0; float l2 = -3.4e38f;
    for (int e = 0; e < E; ++e) {
      if (e == i1) continue;
      float v = ae[gsel * E + e];
      if (v > l2) { l2 = v; i2 = e; }
    }
    float e2 = __expf(l2 - l1);
    float w1 = 1.f / (1.f + e2);
    float w2 = e2 / (1.f + e2);
    float* cn = c + (size_t)n * GE;
    for (int k = 0; k < GE; ++k) cn[k] = 0.f;
    cn[gsel * E + i1] = w1;
    cn[gsel * E + i2] = w2;
  }
}

// ---------------------------------------------------------------------------
// bf16 WMMA GEMM: C = A (M x K, bf16) * Bw^T (Nn x K, fp32 weights cvt->bf16)
// 128x128 tile / 256 threads; 8 waves x (64x32) = 4x2 WMMA tiles each.
// Double-buffered LDS pipeline; A tile via CDNA5 async global->LDS DMA.
// epi 0: Cbf = bf16(relu(acc + bias[col]))
// epi 1: Cf  = rowscale[row]*(acc + bias[col])  (+= when accumulate)
// ---------------------------------------------------------------------------
__global__ __launch_bounds__(256) void moe_gemm_bf16_wmma(
    const __bf16* __restrict__ A,
    const float*  __restrict__ Bw,
    const float*  __restrict__ bias,
    __bf16* __restrict__ Cbf,
    float*  __restrict__ Cf,
    const float* __restrict__ rowscale,
    int M, int Nn, int K,
    int epi, int accumulate, int scale_stride) {
  __shared__ __bf16 As[2][TILE_M * LDS_K];
  __shared__ __bf16 Bs[2][TILE_N * LDS_K];

  const int tid  = threadIdx.x;
  const int lane = tid & 31;
  const int wave = tid >> 5;
  const int wm = wave & 1;   // 2 wave-rows of 64
  const int wn = wave >> 1;  // 4 wave-cols of 32

  const int m0 = blockIdx.y * TILE_M;
  const int n0 = blockIdx.x * TILE_N;

  f32x8 acc[4][2];
#pragma unroll
  for (int i = 0; i < 4; ++i)
#pragma unroll
    for (int j = 0; j < 2; ++j)
#pragma unroll
      for (int r = 0; r < 8; ++r) acc[i][j][r] = 0.f;

  // --- stage helpers -------------------------------------------------------
  // A tile: 128 rows x 32 bf16 = 512 x 16B chunks, 2 per thread (async DMA).
  auto issueA = [&](int buf, int k0) {
#pragma unroll
    for (int it = 0; it < 2; ++it) {
      int ci  = tid + it * 256;
      int row = ci >> 2;
      int kq  = (ci & 3) * 8;
      const __bf16* gp = A + (size_t)(m0 + row) * K + k0 + kq;
      __bf16* lp = &As[buf][row * LDS_K + kq];
#if HAVE_ASYNC_LDS
      __builtin_amdgcn_global_load_async_to_lds_b128(
          (v4i*)gp, (v4i_lds*)lp, 0, 0);
#else
      *(bf16x8*)lp = *(const bf16x8*)gp;
#endif
    }
  };
  // B tile: 128 rows x 32 fp32 = 1024 float4, 4 per thread.
  auto loadBregs = [&](int k0, f32x4 br[4]) {
#pragma unroll
    for (int it = 0; it < 4; ++it) {
      int fi  = tid + it * 256;
      int row = fi >> 3;
      int kq  = (fi & 7) * 4;
      br[it] = *(const f32x4*)(Bw + (size_t)(n0 + row) * K + k0 + kq);
    }
  };
  auto storeB = [&](int buf, const f32x4 br[4]) {
#pragma unroll
    for (int it = 0; it < 4; ++it) {
      int fi  = tid + it * 256;
      int row = fi >> 3;
      int kq  = (fi & 7) * 4;
      bf16x4 o;
      o[0] = (__bf16)br[it][0]; o[1] = (__bf16)br[it][1];
      o[2] = (__bf16)br[it][2]; o[3] = (__bf16)br[it][3];
      *(bf16x4*)(&Bs[buf][row * LDS_K + kq]) = o;
    }
  };
  // One K-stage of WMMAs from LDS buffer `buf`.
  auto compute = [&](int buf) {
    const int khalf = lane >> 4;   // K-half selector per ISA fragment layout
    const int mr    = lane & 15;
    bf16x16 bfrag[2];
#pragma unroll
    for (int j = 0; j < 2; ++j) {
      int nrow = wn * 32 + j * 16 + mr;
      int kb   = khalf * 8;
      bf16x8 lo = *(const bf16x8*)(&Bs[buf][nrow * LDS_K + kb]);
      bf16x8 hi = *(const bf16x8*)(&Bs[buf][nrow * LDS_K + kb + 16]);
      bfrag[j] = __builtin_shufflevector(lo, hi, 0, 1, 2, 3, 4, 5, 6, 7,
                                         8, 9, 10, 11, 12, 13, 14, 15);
    }
#pragma unroll
    for (int i = 0; i < 4; ++i) {
      int ar = wm * 64 + i * 16 + mr;
      bf16x8 lo = *(const bf16x8*)(&As[buf][ar * LDS_K + khalf * 8]);
      bf16x8 hi = *(const bf16x8*)(&As[buf][ar * LDS_K + 16 + khalf * 8]);
      bf16x16 afrag = __builtin_shufflevector(lo, hi, 0, 1, 2, 3, 4, 5, 6, 7,
                                              8, 9, 10, 11, 12, 13, 14, 15);
#pragma unroll
      for (int j = 0; j < 2; ++j) {
        acc[i][j] = __builtin_amdgcn_wmma_f32_16x16x32_bf16(
            false, afrag, false, bfrag[j], (short)0, acc[i][j], false, false);
      }
    }
  };

  // --- double-buffered main pipeline --------------------------------------
  const int nk = K / TILE_K;
  {
    f32x4 br[4];
    issueA(0, 0);
    loadBregs(0, br);
    storeB(0, br);
    wait_async_then_barrier();
  }
  for (int kt = 1; kt < nk; ++kt) {
    const int buf = kt & 1, pbuf = buf ^ 1;
    f32x4 br[4];
    loadBregs(kt * TILE_K, br);      // global loads for stage kt (in regs)
    issueA(buf, kt * TILE_K);        // async DMA for stage kt
    if (kt + 1 < nk) {               // prefetch B panel one stage further out
      int row = tid >> 3, kq = (tid & 7) * 4;
      __builtin_prefetch(Bw + (size_t)(n0 + row) * K + (kt + 1) * TILE_K + kq, 0, 1);
    }
    compute(pbuf);                   // WMMAs on stage kt-1 overlap the loads
    storeB(buf, br);
    wait_async_then_barrier();
  }
  compute((nk - 1) & 1);

  // --- epilogue. C/D layout: VGPR r -> row (lane>>4)*8 + r, col lane&15. ---
  const int rbase = (lane >> 4) * 8;
  const int ccol  = lane & 15;
  if (epi == 0) {
#pragma unroll
    for (int i = 0; i < 4; ++i)
#pragma unroll
      for (int j = 0; j < 2; ++j) {
        int trow = m0 + wm * 64 + i * 16 + rbase;
        int tcol = n0 + wn * 32 + j * 16 + ccol;
        float bcol = bias[tcol];
#pragma unroll
        for (int r = 0; r < 8; ++r) {
          float v = acc[i][j][r] + bcol;
          v = v > 0.f ? v : 0.f;
          Cbf[(size_t)(trow + r) * Nn + tcol] = (__bf16)v;
        }
      }
  } else {
#pragma unroll
    for (int i = 0; i < 4; ++i)
#pragma unroll
      for (int j = 0; j < 2; ++j) {
        int trow = m0 + wm * 64 + i * 16 + rbase;
        int tcol = n0 + wn * 32 + j * 16 + ccol;
        float bcol = bias[tcol];
#pragma unroll
        for (int r = 0; r < 8; ++r) {
          int row = trow + r;
          float s = rowscale[(size_t)row * scale_stride];
          float o = s * (acc[i][j][r] + bcol);
          if (accumulate) o += Cf[(size_t)row * Nn + tcol];
          Cf[(size_t)row * Nn + tcol] = o;
        }
      }
  }
}

// ---------------------------------------------------------------------------
extern "C" void kernel_launch(void* const* d_in, const int* in_sizes, int n_in,
                              void* d_out, int out_size, void* d_ws, size_t ws_size,
                              hipStream_t stream) {
  const float* xs  = (const float*)d_in[0];
  const float* bnk = (const float*)d_in[1];
  const float* Wlg = (const float*)d_in[2];
  const float* blg = (const float*)d_in[3];
  const float* Wg  = (const float*)d_in[4];
  const float* W1  = (const float*)d_in[5];
  const float* b1  = (const float*)d_in[6];
  const float* W2  = (const float*)d_in[7];
  const float* b2  = (const float*)d_in[8];
  // d_in[9] = top_k (fixed 2 by construction; top-2 hardcoded in router)

  const int G = in_sizes[3];
  const int D = in_sizes[2] / G;
  const int N = in_sizes[0] / D;            // B*L tokens
  const int E = in_sizes[4] / (G * D);
  const int H = in_sizes[5] / (G * E * D);

  // workspace layout (256B-aligned blocks)
  char* ws = (char*)d_ws;
  __bf16* x_bf = (__bf16*)ws;                                   // N*D bf16
  size_t off = (((size_t)N * D * 2) + 255) & ~(size_t)255;
  __bf16* h_bf = (__bf16*)(ws + off);                           // N*H bf16 (reused per expert)
  off += ((((size_t)N * H * 2) + 255) & ~(size_t)255);
  float* c = (float*)(ws + off);                                // N*(G*E) f32 coefficients

  float* out = (float*)d_out;

  // 1) x -> bf16
  {
    int n4 = (N * D) / 4;
    moe_cast_bf16<<<dim3((n4 + 255) / 256), dim3(256), 0, stream>>>(xs, x_bf, n4);
  }
  // 2) routing coefficients
  moe_router<<<dim3((N + 7) / 8), dim3(256), 0, stream>>>(xs, bnk, Wlg, blg, Wg, c, N, D, G, E);

  // 3) per (group, expert): weighted dense FFN via two WMMA GEMMs
  dim3 blk(256);
  for (int g = 0; g < G; ++g) {
    for (int e = 0; e < E; ++e) {
      int ge = g * E + e;
      // h = bf16(relu(x @ W1[g,e]^T + b1[g,e]))
      dim3 g1(H / TILE_N, N / TILE_M);
      moe_gemm_bf16_wmma<<<g1, blk, 0, stream>>>(
          x_bf, W1 + (size_t)ge * H * D, b1 + (size_t)ge * H,
          h_bf, nullptr, nullptr,
          N, H, D, /*epi=*/0, /*accumulate=*/0, /*scale_stride=*/0);
      // out (+)= c[:,ge] * (h @ W2[g,e]^T + b2[g,e])
      dim3 g2(D / TILE_N, N / TILE_M);
      moe_gemm_bf16_wmma<<<g2, blk, 0, stream>>>(
          h_bf, W2 + (size_t)ge * D * H, b2 + (size_t)ge * D,
          nullptr, out, c + ge,
          N, D, H, /*epi=*/1, /*accumulate=*/(ge == 0 ? 0 : 1), /*scale_stride=*/G * E);
    }
  }
}
